// MultGran_STGCN_44779329028461
// MI455X (gfx1250) — matrix operand order repaired and checked by reading, not compiled, not run on hardware
//
#include <hip/hip_runtime.h>
#include <math.h>

typedef __attribute__((ext_vector_type(16))) __bf16 v16bf;
typedef __attribute__((ext_vector_type(8)))  float  v8f;

#define BDIM 256

// ---- problem dims (from the reference) ----
#define BB    16
#define LL    12
#define OUTC  12
#define NNODE 512
#define NCLU  64
#define NL3C  16
#define RC    32
#define SKC   256
#define ECC   512

// ============================================================================
// Generic strided GEMM:  C[z][m][j] = sum_k A[z][m][k] * B[z][k][j] (+bias[m])
// j=(j1,j2) inner size n2; k=(k1,k2) inner size k2.  One wave32 computes a
// 32x32 tile: 2 A frags x 2 B frags -> 4 v_wmma_f32_16x16x32_bf16 per K step.
// Fast path (K-contiguous A, full K tiles): A via global_load_b128, no masks
// (out-of-range rows/cols only feed discarded D rows/cols).  Tail/messy path:
// scalar loads masked only on the K bound.  All k/n index decompositions use
// the exact magic-multiply div/mod (branch-free; degenerates to linear offset
// when the decomposition is linear, so no special cases are needed).
// ============================================================================
struct GemmDesc {
  const float* A; const float* Bm; float* C; const float* bias;
  int M, Nd, K, n2, k2;
  int fastA;
  unsigned long long magN2, magK2;   // ceil(2^32 / n2), ceil(2^32 / k2)
  long sAb, sAm, sAk1, sAk2;
  long sBb, sBk1, sBk2, sBn1, sBn2;
  long sCb, sCm, sCn1, sCn2;
};

__device__ __forceinline__ void dm(unsigned x, unsigned dvs, unsigned long long mg,
                                   unsigned& q, unsigned& r) {
  q = (unsigned)((x * mg) >> 32);    // exact for x,dvs <= 2^16
  r = x - q * dvs;
}

__global__ void wmma_gemm_bf16(GemmDesc d) {
  const int lane = threadIdx.x & 31;
  const int wave = threadIdx.x >> 5;
  const int wpb  = blockDim.x >> 5;
  const int tilesN = (d.Nd + 31) >> 5;
  const int tilesM = (d.M + 31) >> 5;
  long tile = (long)blockIdx.x * wpb + wave;
  if (tile >= (long)tilesM * tilesN) return;   // wave-uniform exit: EXEC stays full
  const int tm = (int)(tile / tilesN);
  const int tn = (int)(tile % tilesN);
  const int z  = blockIdx.y;
  const int mBase = tm << 5, nBase = tn << 5;
  const int l15 = lane & 15;
  const bool hi = lane >= 16;

  const int m0 = mBase + l15,      m1 = mBase + 16 + l15;
  const int n0 = nBase + l15,      n1 = nBase + 16 + l15;
  const bool n0ok = n0 < d.Nd, n1ok = n1 < d.Nd;

  const long baseA = d.sAb * z;
  const long baseB = d.sBb * z;

  unsigned q, r;
  dm(n0ok ? (unsigned)n0 : 0u, (unsigned)d.n2, d.magN2, q, r);
  const long offN0  = d.sBn1 * q + d.sBn2 * r;
  const long offCN0 = d.sCn1 * q + d.sCn2 * r;
  dm(n1ok ? (unsigned)n1 : 0u, (unsigned)d.n2, d.magN2, q, r);
  const long offN1  = d.sBn1 * q + d.sBn2 * r;
  const long offCN1 = d.sCn1 * q + d.sCn2 * r;

  const long offM0 = d.sAm * (long)((m0 < d.M) ? m0 : 0);
  const long offM1 = d.sAm * (long)((m1 < d.M) ? m1 : 0);

  v8f acc00 = {0.f,0.f,0.f,0.f,0.f,0.f,0.f,0.f};
  v8f acc01 = {0.f,0.f,0.f,0.f,0.f,0.f,0.f,0.f};
  v8f acc10 = {0.f,0.f,0.f,0.f,0.f,0.f,0.f,0.f};
  v8f acc11 = {0.f,0.f,0.f,0.f,0.f,0.f,0.f,0.f};

  int k0 = 0;
  const int kFull = d.K & ~31;
  if (d.fastA) {
    for (; k0 < kFull; k0 += 32) {   // full tiles, A contiguous in k: b128 loads
      // A 16x32 bf16 layout: lanes 0-15 K{0..7,16..23}, lanes 16-31 K{8..15,24..31}
      const float* ap0 = d.A + baseA + offM0 + k0 + (hi ? 8 : 0);
      const float* ap1 = d.A + baseA + offM1 + k0 + (hi ? 8 : 0);
      float4 x00 = *(const float4*)(ap0);
      float4 x01 = *(const float4*)(ap0 + 4);
      float4 x02 = *(const float4*)(ap0 + 16);
      float4 x03 = *(const float4*)(ap0 + 20);
      float4 x10 = *(const float4*)(ap1);
      float4 x11 = *(const float4*)(ap1 + 4);
      float4 x12 = *(const float4*)(ap1 + 16);
      float4 x13 = *(const float4*)(ap1 + 20);
      float f0[16] = {x00.x,x00.y,x00.z,x00.w, x01.x,x01.y,x01.z,x01.w,
                      x02.x,x02.y,x02.z,x02.w, x03.x,x03.y,x03.z,x03.w};
      float f1[16] = {x10.x,x10.y,x10.z,x10.w, x11.x,x11.y,x11.z,x11.w,
                      x12.x,x12.y,x12.z,x12.w, x13.x,x13.y,x13.z,x13.w};
      v16bf a0, a1, b0, b1;
#pragma unroll
      for (int e = 0; e < 16; ++e) { a0[e] = (__bf16)f0[e]; a1[e] = (__bf16)f1[e]; }
      // B 32x16 bf16 layout: lanes 0-15 K{0..7,16..23}, lanes 16-31 K{8..15,24..31}
#pragma unroll
      for (int e = 0; e < 16; ++e) {
        unsigned kkb = (unsigned)(k0 + (hi ? ((e < 8) ? e + 8 : e + 16) : ((e < 8) ? e : e + 8)));
        dm(kkb, (unsigned)d.k2, d.magK2, q, r);
        long offKB = d.sBk1 * q + d.sBk2 * r;
        b0[e] = (__bf16)d.Bm[baseB + offKB + offN0];
        b1[e] = (__bf16)d.Bm[baseB + offKB + offN1];
      }
      __builtin_prefetch(ap0 + 32, 0, 1);   // next K tile (global_prefetch_b8)
      acc00 = __builtin_amdgcn_wmma_f32_16x16x32_bf16(false, a0, false, b0, (short)0, acc00, false, false);
      acc01 = __builtin_amdgcn_wmma_f32_16x16x32_bf16(false, a0, false, b1, (short)0, acc01, false, false);
      acc10 = __builtin_amdgcn_wmma_f32_16x16x32_bf16(false, a1, false, b0, (short)0, acc10, false, false);
      acc11 = __builtin_amdgcn_wmma_f32_16x16x32_bf16(false, a1, false, b1, (short)0, acc11, false, false);
    }
  }
  for (; k0 < d.K; k0 += 32) {       // generic/tail path: masked only on K bound
    v16bf a0, a1, b0, b1;
#pragma unroll
    for (int e = 0; e < 16; ++e) {
      int ka = hi ? ((e < 8) ? e + 8 : e + 16) : ((e < 8) ? e : e + 8);
      unsigned kk = (unsigned)(k0 + ka);
      bool kok = kk < (unsigned)d.K;
      dm(kok ? kk : 0u, (unsigned)d.k2, d.magK2, q, r);
      long offK = d.sAk1 * q + d.sAk2 * r;
      float va0 = d.A[baseA + offM0 + offK];   // unconditional, in-bounds by clamp
      float va1 = d.A[baseA + offM1 + offK];
      a0[e] = (__bf16)(kok ? va0 : 0.f);
      a1[e] = (__bf16)(kok ? va1 : 0.f);
      unsigned kkb = (unsigned)(k0 + (hi ? ((e < 8) ? e + 8 : e + 16) : ((e < 8) ? e : e + 8)));
      bool kbok = kkb < (unsigned)d.K;
      dm(kbok ? kkb : 0u, (unsigned)d.k2, d.magK2, q, r);
      long offKB = d.sBk1 * q + d.sBk2 * r;
      float vb0 = d.Bm[baseB + offKB + offN0];
      float vb1 = d.Bm[baseB + offKB + offN1];
      b0[e] = (__bf16)(kbok ? vb0 : 0.f);
      b1[e] = (__bf16)(kbok ? vb1 : 0.f);
    }
    acc00 = __builtin_amdgcn_wmma_f32_16x16x32_bf16(false, a0, false, b0, (short)0, acc00, false, false);
    acc01 = __builtin_amdgcn_wmma_f32_16x16x32_bf16(false, a0, false, b1, (short)0, acc01, false, false);
    acc10 = __builtin_amdgcn_wmma_f32_16x16x32_bf16(false, a1, false, b0, (short)0, acc10, false, false);
    acc11 = __builtin_amdgcn_wmma_f32_16x16x32_bf16(false, a1, false, b1, (short)0, acc11, false, false);
  }

  const long baseC = d.sCb * z;
#pragma unroll
  for (int rr = 0; rr < 8; ++rr) {
    // C/D layout: VGPR rr -> M = rr (lanes 0-15) / rr+8 (lanes 16-31), N = lane%16
    int mr0 = mBase + rr + (hi ? 8 : 0);
    int mr1 = mr0 + 16;
    if (mr0 < d.M) {
      float bv = d.bias ? d.bias[mr0] : 0.f;
      long rowOff = baseC + d.sCm * mr0;
      if (n0ok) d.C[rowOff + offCN0] = acc00[rr] + bv;
      if (n1ok) d.C[rowOff + offCN1] = acc01[rr] + bv;
    }
    if (mr1 < d.M) {
      float bv = d.bias ? d.bias[mr1] : 0.f;
      long rowOff = baseC + d.sCm * mr1;
      if (n0ok) d.C[rowOff + offCN0] = acc10[rr] + bv;
      if (n1ok) d.C[rowOff + offCN1] = acc11[rr] + bv;
    }
  }
}

// ============================================================================
// Elementwise / reduction helper kernels
// ============================================================================
__device__ __forceinline__ float blockSum(float v, float* sh) {
  sh[threadIdx.x] = v; __syncthreads();
  for (int s = BDIM >> 1; s > 0; s >>= 1) {
    if (threadIdx.x < s) sh[threadIdx.x] += sh[threadIdx.x + s];
    __syncthreads();
  }
  float r = sh[0]; __syncthreads();
  return r;
}

// BatchNorm2d stats per channel over (B, N*L)
__global__ void k_channel_stats(const float* x, int Bv, int C, int NL,
                                float* mean, float* invstd) {
  int c = blockIdx.x;
  __shared__ float sh[BDIM];
  float a = 0.f, qq = 0.f;
  long total = (long)Bv * NL;
  for (long i = threadIdx.x; i < total; i += blockDim.x) {
    long b = i / NL, r = i % NL;
    float v = x[(b * C + c) * (long)NL + r];
    a += v; qq += v * v;
  }
  float s  = blockSum(a, sh);
  float ss = blockSum(qq, sh);
  if (threadIdx.x == 0) {
    float mn = s / (float)total;
    mean[c] = mn;
    invstd[c] = rsqrtf(ss / (float)total - mn * mn + 1e-5f);
  }
}

__global__ void k_bn_apply(const float* x, float* y, const float* mean,
                           const float* invstd, int C, int NL, long total) {
  long i = (long)blockIdx.x * blockDim.x + threadIdx.x;
  if (i >= total) return;
  int c = (int)((i / NL) % C);
  y[i] = (x[i] - mean[c]) * invstd[c];
}

// A = rownorm(relu(v1 @ v2)), v1:(n,kd) v2:(kd,n); one block per row
__global__ void k_adapt(const float* v1, const float* v2, float* A, int n, int kd) {
  int row = blockIdx.x;
  __shared__ float sh[BDIM];
  float part = 0.f;
  for (int j = threadIdx.x; j < n; j += blockDim.x) {
    float acc = 0.f;
    for (int t = 0; t < kd; ++t) acc += v1[row * kd + t] * v2[t * n + j];
    acc = fmaxf(acc, 0.f);
    A[(long)row * n + j] = acc;
    part += acc;
  }
  float s = blockSum(part, sh);
  float inv = 1.f / s;
  for (int j = threadIdx.x; j < n; j += blockDim.x)
    A[(long)row * n + j] *= inv;
}

// AT[r][c] = A[c][r]  (materialize transposed adjacency so GEMM A is k-contig)
__global__ void k_transpose(const float* A, float* AT, int n, long total) {
  long i = (long)blockIdx.x * blockDim.x + threadIdx.x;
  if (i >= total) return;
  long rr = i / n, cc = i % n;
  AT[i] = A[cc * n + rr];
}

// out[b][c][p] = tanh(h[b][c][p]) * sigmoid(h[b][C+c][p]); h has 2C channels
__global__ void k_gate(const float* h, float* out, long outB, int C, int NT, long total) {
  long i = (long)blockIdx.x * blockDim.x + threadIdx.x;
  if (i >= total) return;
  long per = (long)C * NT;
  long b = i / per, r = i % per;
  float t = tanhf(h[b * 2 * per + r]);
  float g = h[b * 2 * per + per + r];
  g = 1.f / (1.f + __expf(-g));
  out[b * outB + r] = t * g;
}

__global__ void k_sigmoid_bias(float* x, const float* bias, long total, int per) {
  long i = (long)blockIdx.x * blockDim.x + threadIdx.x;
  if (i >= total) return;
  float v = x[i] + bias[i % per];
  x[i] = 1.f / (1.f + __expf(-v));
}

// logits(b,n,m): v = sigmoid(v - mean_n(v)) * tmat[n,m]; one block per (b,m)
__global__ void k_transmit_post(float* T, const float* tmat, int nodes, int m) {
  int blk = blockIdx.x;
  int b = blk / m, mm = blk % m;
  __shared__ float sh[BDIM];
  float* base = T + (long)b * nodes * m;
  float a = 0.f;
  for (int nn = threadIdx.x; nn < nodes; nn += blockDim.x) a += base[(long)nn * m + mm];
  float mean = blockSum(a, sh) / (float)nodes;
  for (int nn = threadIdx.x; nn < nodes; nn += blockDim.x) {
    long off = (long)nn * m + mm;
    float v = base[off] - mean;
    v = 1.f / (1.f + __expf(-v));
    base[off] = v * tmat[off];
  }
}

// S[b,q,p] = softmax_p( t2[b,p,q] )  (t2: (b,l,l), l <= 6)
__global__ void k_softmax_T(const float* t2, float* S, int Bv, int l) {
  int i = blockIdx.x * blockDim.x + threadIdx.x;
  if (i >= Bv * l) return;
  int b = i / l, q = i % l;
  const float* base = t2 + (long)b * l * l + q;
  float mx = -1e30f;
  for (int p = 0; p < l; ++p) mx = fmaxf(mx, base[(long)p * l]);
  float e[8], sum = 0.f;
  for (int p = 0; p < l; ++p) { e[p] = __expf(base[(long)p * l] - mx); sum += e[p]; }
  float* out = S + (long)b * l * l + (long)q * l;
  for (int p = 0; p < l; ++p) out[p] = e[p] / sum;
}

// y = leaky_relu(g,0.01) + res[..., lres-q : ]
__global__ void k_leaky_add(const float* g, const float* res, float* out, long outB,
                            int C, int nodes, int q, int lres, long total) {
  long i = (long)blockIdx.x * blockDim.x + threadIdx.x;
  if (i >= total) return;
  int t = (int)(i % q);
  long rest = i / q;                       // (b*C + c)*nodes + n
  float v = g[i];
  v = (v > 0.f) ? v : 0.01f * v;
  v += res[rest * lres + (lres - q + t)];
  long per = (long)C * nodes * q;
  long b = i / per, within = i % per;
  out[b * outB + within] = v;
}

// LayerNorm([C,N,T]) per batch: contiguous region of `count` at y + b*bStride
__global__ void k_batch_stats(const float* y, long bStride, long count,
                              float* mean, float* invstd) {
  int b = blockIdx.x;
  const float* p = y + (long)b * bStride;
  __shared__ float sh[BDIM];
  float a = 0.f, qq = 0.f;
  for (long i = threadIdx.x; i < count; i += blockDim.x) { float v = p[i]; a += v; qq += v * v; }
  float s  = blockSum(a, sh);
  float ss = blockSum(qq, sh);
  if (threadIdx.x == 0) {
    float mn = s / (float)count;
    mean[b] = mn;
    invstd[b] = rsqrtf(ss / (float)count - mn * mn + 1e-5f);
  }
}

__global__ void k_ln_apply(float* y, long bStride, long count, const float* mean,
                           const float* invstd, const float* g, const float* bia,
                           long total) {
  long i = (long)blockIdx.x * blockDim.x + threadIdx.x;
  if (i >= total) return;
  long b = i / count, r = i % count;
  long off = b * bStride + r;
  y[off] = (y[off] - mean[b]) * invstd[b] * g[r] + bia[r];
}

__global__ void k_relu(float* x, long n) {
  long i = (long)blockIdx.x * blockDim.x + threadIdx.x;
  if (i < n) x[i] = fmaxf(x[i], 0.f);
}

// out = s2 + skip[..., l1-l2 : ]   (layouts (b,C,n,l2) and (b,C,n,l1))
__global__ void k_add_slice(const float* s2, const float* skip, float* out,
                            int l1, int l2, long total) {
  long i = (long)blockIdx.x * blockDim.x + threadIdx.x;
  if (i >= total) return;
  int t = (int)(i % l2);
  long rest = i / l2;
  out[i] = s2[i] + skip[rest * l1 + (l1 - l2 + t)];
}

// out = hr*Wr + hd*Wd + hw*Ww  (W broadcast over batch, per = OUT*N)
__global__ void k_final(const float* hr, const float* hd, const float* hw,
                        const float* Wr, const float* Wd, const float* Ww,
                        float* out, int per, long total) {
  long i = (long)blockIdx.x * blockDim.x + threadIdx.x;
  if (i >= total) return;
  int r = (int)(i % per);
  out[i] = hr[i] * Wr[r] + hd[i] * Wd[r] + hw[i] * Ww[r];
}

// ============================================================================
// Host-side orchestration
// ============================================================================
namespace {

struct Ws {
  char* base; size_t off;
  float* alloc(size_t n) {
    float* p = (float*)(base + off);
    off += ((n * sizeof(float) + 255) & ~(size_t)255);
    return p;
  }
  size_t mark() const { return off; }
  void reset(size_t m) { off = m; }
};

struct PoolP {
  const float *res_w, *res_b, *tc_w, *tc_b, *gcn_w, *gcn_b;
  const float *ta_c1, *ta_c2, *ta_w, *ta_b, *ta_v, *ln_g, *ln_b;
};
struct TrP { const float *c1, *c2, *w, *b; };

inline dim3 ewGrid(long n) { return dim3((unsigned)((n + BDIM - 1) / BDIM)); }

inline unsigned long long magic32(unsigned dvs) {
  return (((unsigned long long)1 << 32) + dvs - 1) / dvs;
}

void gemm_launch(hipStream_t st, GemmDesc d, int batch) {
  d.magK2 = magic32((unsigned)d.k2);
  d.magN2 = magic32((unsigned)d.n2);
  bool aLin = (d.k2 >= d.K) || (d.sAk1 == (long)d.k2 * d.sAk2);
  d.fastA = (aLin && d.sAk2 == 1 && (d.sAm % 4 == 0) && (d.sAb % 4 == 0)) ? 1 : 0;
  int tiles = ((d.M + 31) / 32) * ((d.Nd + 31) / 32);
  int wpb = BDIM / 32;
  dim3 grid((unsigned)((tiles + wpb - 1) / wpb), (unsigned)batch, 1);
  wmma_gemm_bf16<<<grid, dim3(BDIM), 0, st>>>(d);
}

// conv1x1 with optional l-window: reads x[b, c, n, lOff + t], t < lTake
void conv1x1_l(hipStream_t st, const float* w, const float* bias, const float* x,
               long xB, int Cin, int nodes, int lFull, int lOff, int lTake,
               float* out, long outB, int Cout) {
  GemmDesc d{};
  d.A = w; d.Bm = x + lOff; d.C = out; d.bias = bias;
  d.M = Cout; d.Nd = nodes * lTake; d.K = Cin; d.n2 = lTake; d.k2 = Cin;
  d.sAm = Cin; d.sAk2 = 1;
  d.sBb = xB; d.sBk2 = (long)nodes * lFull; d.sBn1 = lFull; d.sBn2 = 1;
  d.sCb = outB; d.sCm = (long)nodes * lTake; d.sCn1 = lTake; d.sCn2 = 1;
  gemm_launch(st, d, BB);
}

// diffusion: out[b,c,m,t] = sum_n h[b,c,n,t] * A[n,m]  (AT = A transposed,
// row-major (m,n) so the GEMM A operand is contiguous in k -> b128 fast path)
void diffusion(hipStream_t st, const float* AT, const float* hin, long hB,
               float* hout, long oB, int nodes, int t) {
  GemmDesc d{};
  d.A = AT; d.Bm = hin; d.C = hout; d.bias = nullptr;
  d.M = nodes; d.Nd = RC * t; d.K = nodes; d.n2 = t; d.k2 = nodes;
  d.sAm = nodes; d.sAk2 = 1;
  d.sBb = hB; d.sBk2 = t; d.sBn1 = (long)nodes * t; d.sBn2 = 1;
  d.sCb = oB; d.sCm = t; d.sCn1 = (long)nodes * t; d.sCn2 = 1;
  gemm_launch(st, d, BB);
}

// _transmit: returns T (b, nodes, m)
float* run_transmit(hipStream_t st, Ws& ws, const float* seq, long seqB, int nodes,
                    int l, const float* seqc, int m, const TrP& tp, const float* tmat) {
  float* f1 = ws.alloc((size_t)BB * nodes * l);            // (b,n,l)
  { GemmDesc d{};
    d.A = tp.c1; d.Bm = seq; d.C = f1;
    d.M = 1; d.Nd = nodes * l; d.K = RC; d.n2 = nodes * l; d.k2 = RC;
    d.sAk2 = 1;
    d.sBb = seqB; d.sBk2 = (long)nodes * l; d.sBn2 = 1;
    d.sCb = (long)nodes * l; d.sCn2 = 1;
    gemm_launch(st, d, BB); }
  float* f2 = ws.alloc((size_t)BB * RC * m);               // (b,c,m)
  { GemmDesc d{};
    d.A = tp.c2; d.Bm = seqc; d.C = f2;
    d.M = 1; d.Nd = RC * m; d.K = l; d.n2 = m; d.k2 = l;
    d.sAk2 = 1;
    d.sBb = (long)RC * m * l; d.sBk2 = 1; d.sBn1 = (long)m * l; d.sBn2 = l;
    d.sCb = (long)RC * m; d.sCn1 = m; d.sCn2 = 1;
    gemm_launch(st, d, BB); }
  float* g1 = ws.alloc((size_t)BB * nodes * RC);           // (b,n,c)
  { GemmDesc d{};
    d.A = f1; d.Bm = tp.w; d.C = g1;
    d.M = nodes; d.Nd = RC; d.K = l; d.n2 = RC; d.k2 = l;
    d.sAb = (long)nodes * l; d.sAm = l; d.sAk2 = 1;
    d.sBk2 = RC; d.sBn2 = 1;
    d.sCb = (long)nodes * RC; d.sCm = RC; d.sCn2 = 1;
    gemm_launch(st, d, BB); }
  float* T = ws.alloc((size_t)BB * nodes * m);             // (b,n,m)
  { GemmDesc d{};
    d.A = g1; d.Bm = f2; d.C = T;
    d.M = nodes; d.Nd = m; d.K = RC; d.n2 = m; d.k2 = RC;
    d.sAb = (long)nodes * RC; d.sAm = RC; d.sAk2 = 1;
    d.sBb = (long)RC * m; d.sBk2 = m; d.sBn2 = 1;
    d.sCb = (long)nodes * m; d.sCm = m; d.sCn2 = 1;
    gemm_launch(st, d, BB); }
  long tot = (long)BB * nodes * m;
  k_sigmoid_bias<<<ewGrid(tot), BDIM, 0, st>>>(T, tp.b, tot, nodes * m);
  k_transmit_post<<<dim3(BB * m), BDIM, 0, st>>>(T, tmat, nodes, m);
  return T;
}

// x' [b,c,M,l] = sum_n T[b,M,n] * seqc[b,c,n,l]  written into a concat slot
void apply_transmit(hipStream_t st, const float* T, const float* seqc, int nodes,
                    int m, int l, float* outBase, long outB) {
  GemmDesc d{};
  d.A = T; d.Bm = seqc; d.C = outBase;
  d.M = nodes; d.Nd = RC * l; d.K = m; d.n2 = l; d.k2 = m;
  d.sAb = (long)nodes * m; d.sAm = m; d.sAk2 = 1;
  d.sBb = (long)RC * m * l; d.sBk2 = l; d.sBn1 = (long)m * l; d.sBn2 = 1;
  d.sCb = outB; d.sCm = l; d.sCn1 = (long)nodes * l; d.sCn2 = 1;
  gemm_launch(st, d, BB);
}

// Full gcn_pool block; writes (b, 32, nodes, t2) at `out` with batch stride outB
void gcnpool(hipStream_t st, Ws& ws, const float* x, int Cin, int nodes, int l_in,
             int Kt, const float* supT0, const float* supT1, const PoolP& pp,
             float* out, long outB) {
  const int C = RC;
  const int t1 = l_in - 2 * (Kt - 1);
  const int t2 = t1 - (Kt - 1);
  size_t mk = ws.mark();
  const long xB = (long)Cin * nodes * l_in;

  // residual conv1x1
  float* res = ws.alloc((size_t)BB * C * nodes * l_in);
  conv1x1_l(st, pp.res_w, pp.res_b, x, xB, Cin, nodes, l_in, 0, l_in,
            res, (long)C * nodes * l_in, C);

  // gated dilated time conv (dil=2)
  float* h2c = ws.alloc((size_t)BB * 2 * C * nodes * t1);
  { GemmDesc d{};
    d.A = pp.tc_w; d.Bm = x; d.C = h2c; d.bias = pp.tc_b;
    d.M = 2 * C; d.Nd = nodes * t1; d.K = Cin * Kt; d.n2 = t1; d.k2 = Kt;
    d.sAm = (long)Cin * Kt; d.sAk1 = Kt; d.sAk2 = 1;
    d.sBb = xB; d.sBk1 = (long)nodes * l_in; d.sBk2 = 2; d.sBn1 = l_in; d.sBn2 = 1;
    d.sCb = (long)2 * C * nodes * t1; d.sCm = (long)nodes * t1; d.sCn1 = t1; d.sCn2 = 1;
    gemm_launch(st, d, BB); }

  // feats = [h, s0h1, s0h2, s1h1, s1h2] in one 5C-channel buffer
  float* feats = ws.alloc((size_t)BB * 5 * C * nodes * t1);
  const long featB = (long)5 * C * nodes * t1;
  { long tot = (long)BB * C * nodes * t1;
    k_gate<<<ewGrid(tot), BDIM, 0, st>>>(h2c, feats, featB, C, nodes * t1, tot); }
  const float* supsT[2] = {supT0, supT1};
  for (int s = 0; s < 2; ++s) {
    float* h1p = feats + (size_t)(1 + 2 * s) * nodes * t1;
    float* h2p = feats + (size_t)(2 + 2 * s) * nodes * t1;
    diffusion(st, supsT[s], feats, featB, h1p, featB, nodes, t1);
    diffusion(st, supsT[s], h1p,   featB, h2p, featB, nodes, t1);
  }

  // gcn time conv (dil=1) + gating
  float* g2c = ws.alloc((size_t)BB * 2 * C * nodes * t2);
  { GemmDesc d{};
    d.A = pp.gcn_w; d.Bm = feats; d.C = g2c; d.bias = pp.gcn_b;
    d.M = 2 * C; d.Nd = nodes * t2; d.K = 5 * C * Kt; d.n2 = t2; d.k2 = Kt;
    d.sAm = (long)5 * C * Kt; d.sAk1 = Kt; d.sAk2 = 1;
    d.sBb = featB; d.sBk1 = (long)nodes * t1; d.sBk2 = 1; d.sBn1 = t1; d.sBn2 = 1;
    d.sCb = (long)2 * C * nodes * t2; d.sCm = (long)nodes * t2; d.sCn1 = t2; d.sCn2 = 1;
    gemm_launch(st, d, BB); }
  float* g = ws.alloc((size_t)BB * C * nodes * t2);
  { long tot = (long)BB * C * nodes * t2;
    k_gate<<<ewGrid(tot), BDIM, 0, st>>>(g2c, g, (long)C * nodes * t2, C, nodes * t2, tot); }

  // temporal attention TATT_1
  float* f1t = ws.alloc((size_t)BB * t2 * nodes);          // (b,l,n)
  { GemmDesc d{};
    d.A = pp.ta_c1; d.Bm = g; d.C = f1t;
    d.M = 1; d.Nd = t2 * nodes; d.K = C; d.n2 = nodes; d.k2 = C;
    d.sAk2 = 1;
    d.sBb = (long)C * nodes * t2; d.sBk2 = (long)nodes * t2; d.sBn1 = 1; d.sBn2 = t2;
    d.sCb = (long)t2 * nodes; d.sCn1 = nodes; d.sCn2 = 1;
    gemm_launch(st, d, BB); }
  float* f2t = ws.alloc((size_t)BB * C * t2);              // (b,c,l)
  { GemmDesc d{};
    d.A = pp.ta_c2; d.Bm = g; d.C = f2t;
    d.M = 1; d.Nd = C * t2; d.K = nodes; d.n2 = t2; d.k2 = nodes;
    d.sAk2 = 1;
    d.sBb = (long)C * nodes * t2; d.sBk2 = t2; d.sBn1 = (long)nodes * t2; d.sBn2 = 1;
    d.sCb = (long)C * t2; d.sCn1 = t2; d.sCn2 = 1;
    gemm_launch(st, d, BB); }
  float* tmp = ws.alloc((size_t)BB * t2 * C);              // (b,l,c)
  { GemmDesc d{};
    d.A = f1t; d.Bm = pp.ta_w; d.C = tmp;
    d.M = t2; d.Nd = C; d.K = nodes; d.n2 = C; d.k2 = nodes;
    d.sAb = (long)t2 * nodes; d.sAm = nodes; d.sAk2 = 1;
    d.sBk2 = C; d.sBn2 = 1;
    d.sCb = (long)t2 * C; d.sCm = C; d.sCn2 = 1;
    gemm_launch(st, d, BB); }
  float* raw = ws.alloc((size_t)BB * t2 * t2);             // (b,l,q)
  { GemmDesc d{};
    d.A = tmp; d.Bm = f2t; d.C = raw;
    d.M = t2; d.Nd = t2; d.K = C; d.n2 = t2; d.k2 = C;
    d.sAb = (long)t2 * C; d.sAm = C; d.sAk2 = 1;
    d.sBb = (long)C * t2; d.sBk2 = t2; d.sBn2 = 1;
    d.sCb = (long)t2 * t2; d.sCm = t2; d.sCn2 = 1;
    gemm_launch(st, d, BB); }
  { long tot = (long)BB * t2 * t2;
    k_sigmoid_bias<<<ewGrid(tot), BDIM, 0, st>>>(raw, pp.ta_b, tot, t2 * t2); }
  float* t2m = ws.alloc((size_t)BB * t2 * t2);             // (b,p,q)
  { GemmDesc d{};
    d.A = pp.ta_v; d.Bm = raw; d.C = t2m;
    d.M = t2; d.Nd = t2; d.K = t2; d.n2 = t2; d.k2 = t2;
    d.sAm = t2; d.sAk2 = 1;
    d.sBb = (long)t2 * t2; d.sBk2 = t2; d.sBn2 = 1;
    d.sCb = (long)t2 * t2; d.sCm = t2; d.sCn2 = 1;
    gemm_launch(st, d, BB); }
  float* S = ws.alloc((size_t)BB * t2 * t2);               // (b,q,p) softmaxed
  k_softmax_T<<<ewGrid((long)BB * t2), BDIM, 0, st>>>(t2m, S, BB, t2);
  float* gq = ws.alloc((size_t)BB * C * nodes * t2);       // g @ T
  { GemmDesc d{};
    d.A = g; d.Bm = S; d.C = gq;
    d.M = C * nodes; d.Nd = t2; d.K = t2; d.n2 = t2; d.k2 = t2;
    d.sAb = (long)C * nodes * t2; d.sAm = t2; d.sAk2 = 1;
    d.sBb = (long)t2 * t2; d.sBk2 = 1; d.sBn2 = t2;       // B[k=l][j=q] = S[b,q,l]
    d.sCb = (long)C * nodes * t2; d.sCm = t2; d.sCn2 = 1;
    gemm_launch(st, d, BB); }

  // leaky_relu + residual slice, then LayerNorm([C,N,T]) per batch
  { long tot = (long)BB * C * nodes * t2;
    k_leaky_add<<<ewGrid(tot), BDIM, 0, st>>>(gq, res, out, outB, C, nodes, t2, l_in, tot); }
  float* mv = ws.alloc(2 * BB);
  long count = (long)C * nodes * t2;
  k_batch_stats<<<dim3(BB), BDIM, 0, st>>>(out, outB, count, mv, mv + BB);
  { long tot = (long)BB * count;
    k_ln_apply<<<ewGrid(tot), BDIM, 0, st>>>(out, outB, count, mv, mv + BB,
                                             pp.ln_g, pp.ln_b, tot); }
  ws.reset(mk);
}

float* bn(hipStream_t st, Ws& ws, const float* x, int C, int NL) {
  float* y = ws.alloc((size_t)BB * C * NL);
  float* mv = ws.alloc(2 * (size_t)C);
  k_channel_stats<<<dim3(C), BDIM, 0, st>>>(x, BB, C, NL, mv, mv + C);
  long tot = (long)BB * C * NL;
  k_bn_apply<<<ewGrid(tot), BDIM, 0, st>>>(x, y, mv, mv + C, C, NL, tot);
  return y;
}

void head(hipStream_t st, Ws& ws, float* skip, const float* e1w, const float* e1b,
          const float* e2w, const float* e2b, float* outH) {
  size_t mk = ws.mark();
  long tot = (long)BB * SKC * NNODE * 3;
  k_relu<<<ewGrid(tot), BDIM, 0, st>>>(skip, tot);
  float* h1 = ws.alloc((size_t)BB * ECC * NNODE);
  { GemmDesc d{};                                  // end1: (1,3) conv, 3 -> 1
    d.A = e1w; d.Bm = skip; d.C = h1; d.bias = e1b;
    d.M = ECC; d.Nd = NNODE; d.K = SKC * 3; d.n2 = NNODE; d.k2 = 3;
    d.sAm = (long)SKC * 3; d.sAk1 = 3; d.sAk2 = 1;
    d.sBb = (long)SKC * NNODE * 3; d.sBk1 = (long)NNODE * 3; d.sBk2 = 1; d.sBn2 = 3;
    d.sCb = (long)ECC * NNODE; d.sCm = NNODE; d.sCn2 = 1;
    gemm_launch(st, d, BB); }
  long tot2 = (long)BB * ECC * NNODE;
  k_relu<<<ewGrid(tot2), BDIM, 0, st>>>(h1, tot2);
  conv1x1_l(st, e2w, e2b, h1, (long)ECC * NNODE, ECC, NNODE, 1, 0, 1,
            outH, (long)OUTC * NNODE, OUTC);
  ws.reset(mk);
}

} // namespace

extern "C" void kernel_launch(void* const* d_in, const int* in_sizes, int n_in,
                              void* d_out, int out_size, void* d_ws, size_t ws_size,
                              hipStream_t stream) {
  (void)in_sizes; (void)n_in; (void)out_size; (void)ws_size;
  int ii = 0;
  auto nx = [&]() { return (const float*)d_in[ii++]; };

  const float* in_recent = nx();
  const float* in_day    = nx();
  const float* in_week   = nx();
  const float* in_rc     = nx();
  nx(); nx();                                   // day/week cluster: unused
  const float* in_rl3    = nx();
  nx(); nx();                                   // day/week l3: unused

  // params in _make_params insertion order
  const float* nodevec1    = nx(); const float* nodevec2    = nx();
  const float* nodevec1_c  = nx(); const float* nodevec2_c  = nx();
  const float* nodevec1_l3 = nx(); const float* nodevec2_l3 = nx();
  const float* support     = nx(); const float* support_c   = nx();
  const float* support_l3  = nx();
  const float* tmat        = nx(); const float* l3_tmat     = nx();
  const float* start_w   = nx(); const float* start_b   = nx();
  const float* startc_w  = nx(); const float* startc_b  = nx();
  const float* startl3_w = nx(); const float* startl3_b = nx();
  auto rdPool = [&](PoolP& p) {
    p.res_w = nx(); p.res_b = nx(); p.tc_w = nx(); p.tc_b = nx();
    p.gcn_w = nx(); p.gcn_b = nx(); p.ta_c1 = nx(); p.ta_c2 = nx();
    p.ta_w = nx(); p.ta_b = nx(); p.ta_v = nx(); p.ln_g = nx(); p.ln_b = nx();
  };
  PoolP b7, b8, b3, b4, bc1, bc2, bl31, bl32;
  rdPool(b7); rdPool(b8); rdPool(b3); rdPool(b4);
  rdPool(bc1); rdPool(bc2); rdPool(bl31); rdPool(bl32);
  auto rdTr = [&](TrP& t) { t.c1 = nx(); t.c2 = nx(); t.w = nx(); t.b = nx(); };
  TrP tr1, tr2, tr3, tr4, tr5, tr6;
  rdTr(tr1); rdTr(tr2); rdTr(tr3); rdTr(tr4); rdTr(tr5); rdTr(tr6);
  const float* skip1_w = nx(); const float* skip1_b = nx();
  const float* skip2_w = nx(); const float* skip2_b = nx();
  const float* skip3_w = nx(); const float* skip3_b = nx();
  const float* skip4_w = nx(); const float* skip4_b = nx();
  const float* end1_w = nx(); const float* end1_b = nx();
  const float* end2_w = nx(); const float* end2_b = nx();
  const float* W_recent = nx(); const float* W_day = nx(); const float* W_week = nx();

  Ws ws{(char*)d_ws, 0};
  hipStream_t st = stream;

  // ---- BatchNorm (training stats) on the five used inputs ----
  const float* rx_bn   = bn(st, ws, in_recent, 1, NNODE * LL);
  const float* dx_bn   = bn(st, ws, in_day,    1, NNODE * LL);
  const float* wx_bn   = bn(st, ws, in_week,   1, NNODE * LL);
  const float* rxc_bn  = bn(st, ws, in_rc,     3, NCLU  * LL);
  const float* rxl3_bn = bn(st, ws, in_rl3,    2, NL3C  * LL);

  // ---- adaptive supports + transposed copies (k-contiguous GEMM A operand) ----
  float* Aad  = ws.alloc((size_t)NNODE * NNODE);
  float* AadC = ws.alloc((size_t)NCLU * NCLU);
  float* AadL = ws.alloc((size_t)NL3C * NL3C);
  k_adapt<<<dim3(NNODE), BDIM, 0, st>>>(nodevec1,    nodevec2,    Aad,  NNODE, 10);
  k_adapt<<<dim3(NCLU),  BDIM, 0, st>>>(nodevec1_c,  nodevec2_c,  AadC, NCLU,  10);
  k_adapt<<<dim3(NL3C),  BDIM, 0, st>>>(nodevec1_l3, nodevec2_l3, AadL, NL3C,  10);
  float* supT  = ws.alloc((size_t)NNODE * NNODE);
  float* supCT = ws.alloc((size_t)NCLU * NCLU);
  float* supLT = ws.alloc((size_t)NL3C * NL3C);
  float* AadT  = ws.alloc((size_t)NNODE * NNODE);
  float* AadCT = ws.alloc((size_t)NCLU * NCLU);
  float* AadLT = ws.alloc((size_t)NL3C * NL3C);
  k_transpose<<<ewGrid((long)NNODE * NNODE), BDIM, 0, st>>>(support,    supT,  NNODE, (long)NNODE * NNODE);
  k_transpose<<<ewGrid((long)NCLU * NCLU),   BDIM, 0, st>>>(support_c,  supCT, NCLU,  (long)NCLU * NCLU);
  k_transpose<<<ewGrid((long)NL3C * NL3C),   BDIM, 0, st>>>(support_l3, supLT, NL3C,  (long)NL3C * NL3C);
  k_transpose<<<ewGrid((long)NNODE * NNODE), BDIM, 0, st>>>(Aad,  AadT,  NNODE, (long)NNODE * NNODE);
  k_transpose<<<ewGrid((long)NCLU * NCLU),   BDIM, 0, st>>>(AadC, AadCT, NCLU,  (long)NCLU * NCLU);
  k_transpose<<<ewGrid((long)NL3C * NL3C),   BDIM, 0, st>>>(AadL, AadLT, NL3C,  (long)NL3C * NL3C);

  // ---- start convs (rx goes into ch 0-31 of the 96-ch gate4 buffer) ----
  float* rx96 = ws.alloc((size_t)BB * 96 * NNODE * LL);
  const long rx96B = (long)96 * NNODE * LL;
  conv1x1_l(st, start_w, start_b, rx_bn, (long)NNODE * LL, 1, NNODE, LL, 0, LL,
            rx96, rx96B, RC);
  float* dxs = ws.alloc((size_t)BB * RC * NNODE * LL);
  float* wxs = ws.alloc((size_t)BB * RC * NNODE * LL);
  conv1x1_l(st, start_w, start_b, dx_bn, (long)NNODE * LL, 1, NNODE, LL, 0, LL,
            dxs, (long)RC * NNODE * LL, RC);
  conv1x1_l(st, start_w, start_b, wx_bn, (long)NNODE * LL, 1, NNODE, LL, 0, LL,
            wxs, (long)RC * NNODE * LL, RC);
  float* rxcs  = ws.alloc((size_t)BB * RC * NCLU * LL);
  float* rxl3s = ws.alloc((size_t)BB * RC * NL3C * LL);
  conv1x1_l(st, startc_w,  startc_b,  rxc_bn,  (long)3 * NCLU * LL, 3, NCLU, LL, 0, LL,
            rxcs,  (long)RC * NCLU * LL, RC);
  conv1x1_l(st, startl3_w, startl3_b, rxl3_bn, (long)2 * NL3C * LL, 2, NL3C, LL, 0, LL,
            rxl3s, (long)RC * NL3C * LL, RC);

  // ---- gate4: rx = [rx | x1 | x4] ----
  { size_t mk = ws.mark();
    float* T4 = run_transmit(st, ws, rx96, rx96B, NNODE, LL, rxl3s, NL3C, tr4, l3_tmat);
    apply_transmit(st, T4, rxl3s, NNODE, NL3C, LL, rx96 + (size_t)64 * NNODE * LL, rx96B);
    float* T1 = run_transmit(st, ws, rx96, rx96B, NNODE, LL, rxcs, NCLU, tr1, tmat);
    apply_transmit(st, T1, rxcs, NNODE, NCLU, LL, rx96 + (size_t)32 * NNODE * LL, rx96B);
    ws.reset(mk); }

  // ---- first gcn_pool layer (Kt=3, 12 -> 6) ----
  float* rxl3p = ws.alloc((size_t)BB * RC * NL3C * 6);
  float* rxcp  = ws.alloc((size_t)BB * RC * NCLU * 6);
  float* rx96b = ws.alloc((size_t)BB * 96 * NNODE * 6);
  const long rx96bB = (long)96 * NNODE * 6;
  float* dxp = ws.alloc((size_t)BB * RC * NNODE * 6);
  float* wxp = ws.alloc((size_t)BB * RC * NNODE * 6);
  gcnpool(st, ws, rxl3s, RC, NL3C, LL, 3, supLT, AadLT, bl31, rxl3p, (long)RC * NL3C * 6);
  gcnpool(st, ws, rxcs,  RC, NCLU, LL, 3, supCT, AadCT, bc1,  rxcp,  (long)RC * NCLU * 6);
  gcnpool(st, ws, rx96,  96, NNODE, LL, 3, supT,  AadT,  b7,   rx96b, rx96bB);
  gcnpool(st, ws, dxs,   RC, NNODE, LL, 3, supT,  AadT,  b3,   dxp,   (long)RC * NNODE * 6);
  gcnpool(st, ws, wxs,   RC, NNODE, LL, 3, supT,  AadT,  b3,   wxp,   (long)RC * NNODE * 6);

  // ---- gate5: rx = [rx | x5 | x2] ----
  { size_t mk = ws.mark();
    float* T5 = run_transmit(st, ws, rx96b, rx96bB, NNODE, 6, rxl3p, NL3C, tr5, l3_tmat);
    apply_transmit(st, T5, rxl3p, NNODE, NL3C, 6, rx96b + (size_t)32 * NNODE * 6, rx96bB);
    float* T2 = run_transmit(st, ws, rx96b, rx96bB, NNODE, 6, rxcp, NCLU, tr2, tmat);
    apply_transmit(st, T2, rxcp, NNODE, NCLU, 6, rx96b + (size_t)64 * NNODE * 6, rx96bB);
    ws.reset(mk); }

  // ---- skip convs (only the last 3 time steps are ever used) ----
  float* skipR = ws.alloc((size_t)BB * SKC * NNODE * 3);
  float* skipD = ws.alloc((size_t)BB * SKC * NNODE * 3);
  float* skipW = ws.alloc((size_t)BB * SKC * NNODE * 3);
  conv1x1_l(st, skip1_w, skip1_b, rx96b, rx96bB, 96, NNODE, 6, 3, 3,
            skipR, (long)SKC * NNODE * 3, SKC);
  conv1x1_l(st, skip3_w, skip3_b, dxp, (long)RC * NNODE * 6, RC, NNODE, 6, 3, 3,
            skipD, (long)SKC * NNODE * 3, SKC);
  conv1x1_l(st, skip3_w, skip3_b, wxp, (long)RC * NNODE * 6, RC, NNODE, 6, 3, 3,
            skipW, (long)SKC * NNODE * 3, SKC);

  // ---- second gcn_pool layer (Kt=2, 6 -> 3) ----
  float* rxl3p2 = ws.alloc((size_t)BB * RC * NL3C * 3);
  float* rxcp2  = ws.alloc((size_t)BB * RC * NCLU * 3);
  float* rx96c  = ws.alloc((size_t)BB * 96 * NNODE * 3);
  const long rx96cB = (long)96 * NNODE * 3;
  float* dxp2 = ws.alloc((size_t)BB * RC * NNODE * 3);
  float* wxp2 = ws.alloc((size_t)BB * RC * NNODE * 3);
  gcnpool(st, ws, rxl3p, RC, NL3C, 6, 2, supLT, AadLT, bl32, rxl3p2, (long)RC * NL3C * 3);
  gcnpool(st, ws, rxcp,  RC, NCLU, 6, 2, supCT, AadCT, bc2,  rxcp2,  (long)RC * NCLU * 3);
  gcnpool(st, ws, rx96b, 96, NNODE, 6, 2, supT,  AadT,  b8,   rx96c,  rx96cB);
  gcnpool(st, ws, dxp,   RC, NNODE, 6, 2, supT,  AadT,  b4,   dxp2,   (long)RC * NNODE * 3);
  gcnpool(st, ws, wxp,   RC, NNODE, 6, 2, supT,  AadT,  b4,   wxp2,   (long)RC * NNODE * 3);

  // ---- gate6: rx = [rx | x6 | x3] ----
  { size_t mk = ws.mark();
    float* T6 = run_transmit(st, ws, rx96c, rx96cB, NNODE, 3, rxl3p2, NL3C, tr6, l3_tmat);
    apply_transmit(st, T6, rxl3p2, NNODE, NL3C, 3, rx96c + (size_t)32 * NNODE * 3, rx96cB);
    float* T3 = run_transmit(st, ws, rx96c, rx96cB, NNODE, 3, rxcp2, NCLU, tr3, tmat);
    apply_transmit(st, T3, rxcp2, NNODE, NCLU, 3, rx96c + (size_t)64 * NNODE * 3, rx96cB);
    ws.reset(mk); }

  // ---- skip2/skip4 + residual skip accumulation ----
  float* s2r = ws.alloc((size_t)BB * SKC * NNODE * 3);
  float* s2d = ws.alloc((size_t)BB * SKC * NNODE * 3);
  float* s2w = ws.alloc((size_t)BB * SKC * NNODE * 3);
  conv1x1_l(st, skip2_w, skip2_b, rx96c, rx96cB, 96, NNODE, 3, 0, 3,
            s2r, (long)SKC * NNODE * 3, SKC);
  conv1x1_l(st, skip4_w, skip4_b, dxp2, (long)RC * NNODE * 3, RC, NNODE, 3, 0, 3,
            s2d, (long)SKC * NNODE * 3, SKC);
  conv1x1_l(st, skip4_w, skip4_b, wxp2, (long)RC * NNODE * 3, RC, NNODE, 3, 0, 3,
            s2w, (long)SKC * NNODE * 3, SKC);
  { long tot = (long)BB * SKC * NNODE * 3;
    k_add_slice<<<ewGrid(tot), BDIM, 0, st>>>(s2r, skipR, s2r, 3, 3, tot);
    k_add_slice<<<ewGrid(tot), BDIM, 0, st>>>(s2d, skipD, s2d, 3, 3, tot);
    k_add_slice<<<ewGrid(tot), BDIM, 0, st>>>(s2w, skipW, s2w, 3, 3, tot); }

  // ---- output heads and weighted fusion ----
  float* outR = ws.alloc((size_t)BB * OUTC * NNODE);
  float* outD = ws.alloc((size_t)BB * OUTC * NNODE);
  float* outW = ws.alloc((size_t)BB * OUTC * NNODE);
  head(st, ws, s2r, end1_w, end1_b, end2_w, end2_b, outR);
  head(st, ws, s2d, end1_w, end1_b, end2_w, end2_b, outD);
  head(st, ws, s2w, end1_w, end1_b, end2_w, end2_b, outW);
  { long tot = (long)BB * OUTC * NNODE;
    k_final<<<ewGrid(tot), BDIM, 0, st>>>(outR, outD, outW, W_recent, W_day, W_week,
                                          (float*)d_out, OUTC * NNODE, tot); }
}
